// GRU_40003325395616
// MI455X (gfx1250) — matrix operand order repaired and checked by reading, not compiled, not run on hardware
//
#include <hip/hip_runtime.h>
#include <hip/hip_bf16.h>

// ---------------------------------------------------------------------------
// 2-layer GRU, B=32, T=512, I=512, H=1024, output = h1 at t=T-1  ([32,1024] f32)
//
// gfx1250 strategy:
//   * time-parallel input GEMMs (bf16 WMMA, LDS double-buffered via
//     GLOBAL_LOAD_ASYNC_TO_LDS_B128 + s_wait_asynccnt)
//   * fused recurrent step kernel: hprev staged to LDS once per block (async),
//     Wh B-fragments register double-buffered across the K loop,
//     gates r/z/n kept register-resident, sigmoid/tanh + h update fused.
// ---------------------------------------------------------------------------

typedef __attribute__((ext_vector_type(16))) __bf16 v16bf;
typedef __attribute__((ext_vector_type(8)))  __bf16 v8bf;
typedef __attribute__((ext_vector_type(8)))  float  v8f;

constexpr int BB = 32;    // batch
constexpr int TT = 512;   // time
constexpr int II = 512;   // input size
constexpr int HH = 1024;  // hidden
constexpr int GG = 3 * HH;

// ---- workspace layout (bytes) ----------------------------------------------
constexpr size_t OFF_GX   = 0;                                   // f32 [T*B, G]
constexpr size_t SZ_GX    = (size_t)TT * BB * GG * 4;            // 192 MB
constexpr size_t OFF_HALL = OFF_GX + SZ_GX;                      // bf16 [T*B, H]
constexpr size_t SZ_HALL  = (size_t)TT * BB * HH * 2;            // 32 MB
constexpr size_t OFF_XBF  = OFF_HALL + SZ_HALL;                  // bf16 [T*B, I]
constexpr size_t SZ_XBF   = (size_t)TT * BB * II * 2;            // 16 MB
constexpr size_t OFF_WXT0 = OFF_XBF + SZ_XBF;                    // bf16 [G, I]
constexpr size_t SZ_WXT0  = (size_t)GG * II * 2;
constexpr size_t OFF_WHT0 = OFF_WXT0 + SZ_WXT0;                  // bf16 [G, H]
constexpr size_t SZ_WT_H  = (size_t)GG * HH * 2;
constexpr size_t OFF_WXT1 = OFF_WHT0 + SZ_WT_H;                  // bf16 [G, H]
constexpr size_t OFF_WHT1 = OFF_WXT1 + SZ_WT_H;                  // bf16 [G, H]
constexpr size_t OFF_STATE= OFF_WHT1 + SZ_WT_H;
constexpr size_t OFF_HZERO = OFF_STATE;                          // bf16 [B,H]
constexpr size_t SZ_HBF    = (size_t)BB * HH * 2;
constexpr size_t OFF_H0F   = OFF_HZERO + SZ_HBF;                 // f32 [2][B,H]
constexpr size_t SZ_HF     = (size_t)BB * HH * 4;
constexpr size_t OFF_H1BF  = OFF_H0F + 2 * SZ_HF;                // bf16 [2][B,H]
constexpr size_t OFF_H1F   = OFF_H1BF + 2 * SZ_HBF;              // f32 [2][B,H]
constexpr size_t STATE_BYTES = (OFF_H1F + 2 * SZ_HF) - OFF_STATE;

// ---------------------------------------------------------------------------
__device__ __forceinline__ unsigned short f32_to_bf16_rne(float f) {
  unsigned int u = __float_as_uint(f);
  unsigned int r = u + 0x7FFFu + ((u >> 16) & 1u);
  return (unsigned short)(r >> 16);
}
__device__ __forceinline__ float sigmoidf_(float x) {
  return 1.0f / (1.0f + __expf(-x));
}
__device__ __forceinline__ v16bf make_frag(const __bf16* p0, const __bf16* p1) {
  v8bf lo = *(const v8bf*)p0;
  v8bf hi = *(const v8bf*)p1;
  v16bf r;
#pragma unroll
  for (int i = 0; i < 8; ++i) { r[i] = lo[i]; r[8 + i] = hi[i]; }
  return r;
}

// --- gfx1250 async global->LDS copy (per-lane 16B), tracked on ASYNCcnt -----
// LDS flat aperture: LDS byte offset = low 32 bits of the generic pointer.
__device__ __forceinline__ unsigned int lds_off_of(const void* p) {
  return (unsigned int)(unsigned long long)(uintptr_t)p;
}
__device__ __forceinline__ void async_copy_b128(const void* lds_dst,
                                                const void* gsrc) {
  unsigned int       l = lds_off_of(lds_dst);
  unsigned long long g = (unsigned long long)(uintptr_t)gsrc;
  asm volatile("global_load_async_to_lds_b128 %0, %1, off"
               :: "v"(l), "v"(g) : "memory");
}
__device__ __forceinline__ void wait_async0() {
  asm volatile("s_wait_asynccnt 0x0" ::: "memory");
}

// ---------------------------------------------------------------------------
// Transpose + convert f32 [K,N] -> bf16 [N,K]  (32x32 tiles via LDS)
__global__ __launch_bounds__(256) void transpose_cvt_bf16(
    const float* __restrict__ W, unsigned short* __restrict__ WT, int K, int N) {
  __shared__ float tile[32][33];
  const int kb = blockIdx.y * 32, nb = blockIdx.x * 32;
  const int tx = threadIdx.x & 31, ty = threadIdx.x >> 5;
#pragma unroll
  for (int r = ty; r < 32; r += 8)
    tile[r][tx] = W[(size_t)(kb + r) * N + nb + tx];
  __syncthreads();
#pragma unroll
  for (int r = ty; r < 32; r += 8)
    WT[(size_t)(nb + r) * K + kb + tx] = f32_to_bf16_rne(tile[tx][r]);
}

// x [B,T,I] f32 (batch-major) -> bf16 time-major [T*B, I]
__global__ __launch_bounds__(256) void cvt_x_timemajor(
    const float* __restrict__ x, unsigned short* __restrict__ xb) {
  size_t idx = (size_t)blockIdx.x * 256 + threadIdx.x;
  const int i = (int)(idx % II);
  const size_t row = idx / II;                 // = t*B + b
  const int t = (int)(row / BB);
  const int b = (int)(row % BB);
  xb[idx] = f32_to_bf16_rne(x[((size_t)b * TT + t) * II + i]);
}

// ---------------------------------------------------------------------------
// C[M,N] = A[M,K] @ B + bias ; BT[N,K] bf16. Block tile 128x128, async
// double-buffered LDS staging, 8 waves of 64x32.
#define TK 32
#define LDSW 40

__global__ __launch_bounds__(256) void gemm_bf16_bias(
    const __bf16* __restrict__ A, const __bf16* __restrict__ BT,
    const float* __restrict__ bias, float* __restrict__ C,
    int M, int N, int K) {
  __shared__ __attribute__((aligned(16))) __bf16 As[2][128][LDSW];
  __shared__ __attribute__((aligned(16))) __bf16 Bs[2][128][LDSW];
  const int tid  = threadIdx.x;
  const int m0   = blockIdx.y * 128;
  const int n0   = blockIdx.x * 128;
  const int wave = tid >> 5;
  const int lane = tid & 31;
  const int wm   = wave >> 2;
  const int wn   = wave & 3;

  v8f acc[4][2] = {};

  auto loadTile = [&](int buf, int kc) {
#pragma unroll
    for (int i = 0; i < 2; ++i) {
      int c = tid + i * 256;                 // 512 x 16B chunks per matrix
      int row = c >> 2, q = c & 3;
      async_copy_b128(&As[buf][row][q * 8],
                      A + (size_t)(m0 + row) * K + kc + q * 8);
      async_copy_b128(&Bs[buf][row][q * 8],
                      BT + (size_t)(n0 + row) * K + kc + q * 8);
    }
  };

  loadTile(0, 0);
  wait_async0();
  __syncthreads();

  const int nk = K / TK;
  const int rowA = lane & 15, halfA = (lane < 16) ? 0 : 8;
  const int rowB = lane & 15, kB    = (lane < 16) ? 0 : 8;

  for (int kk = 0; kk < nk; ++kk) {
    const int buf = kk & 1;
    if (kk + 1 < nk) loadTile(buf ^ 1, (kk + 1) * TK);

    v16bf afrag[4];
#pragma unroll
    for (int mt = 0; mt < 4; ++mt) {
      const __bf16* ap = &As[buf][wm * 64 + mt * 16 + rowA][0];
      afrag[mt] = make_frag(ap + halfA, ap + 16 + halfA);
    }
    v16bf bfrag[2];
#pragma unroll
    for (int nt = 0; nt < 2; ++nt) {
      const __bf16* bp = &Bs[buf][wn * 32 + nt * 16 + rowB][kB];
      bfrag[nt] = make_frag(bp, bp + 16);
    }
#pragma unroll
    for (int mt = 0; mt < 4; ++mt)
#pragma unroll
      for (int nt = 0; nt < 2; ++nt)
        acc[mt][nt] = __builtin_amdgcn_wmma_f32_16x16x32_bf16(
            false, afrag[mt], false, bfrag[nt], (short)0, acc[mt][nt],
            false, false);
    wait_async0();       // this wave's async writes to buf^1 are in LDS
    __syncthreads();     // everyone done reading buf / writing buf^1
  }

#pragma unroll
  for (int mt = 0; mt < 4; ++mt)
#pragma unroll
    for (int nt = 0; nt < 2; ++nt) {
      const int col   = n0 + wn * 32 + nt * 16 + (lane & 15);
      const float bv  = bias ? bias[col] : 0.0f;
      const int rbase = m0 + wm * 64 + mt * 16 + ((lane < 16) ? 0 : 8);
      v8f a = acc[mt][nt];
#pragma unroll
      for (int r = 0; r < 8; ++r)
        C[(size_t)(rbase + r) * N + col] = a[r] + bv;
    }
}

// ---------------------------------------------------------------------------
// Fused GRU recurrent step. hprev (32x1024 bf16) is async-staged into LDS
// once per block (shared by all 8 waves); Wh B-fragments are register
// double-buffered across the K loop. grid = 8 blocks x 256 thr = 64 strips.
#define APAD 1032  // 1024 + 8 bf16 pad: row stride 2064B -> conflict-free b128

__global__ __launch_bounds__(256) void gru_step(
    const float* __restrict__ gx,            // [B, 3H] slab for this t
    const __bf16* __restrict__ hprev_bf,     // [B, H]
    const float* __restrict__ hprev_f,       // [B, H]
    const __bf16* __restrict__ WhT,          // [3H, H] bf16 transposed
    float* __restrict__ hnew_f,              // [B, H]
    unsigned short* __restrict__ hnew_bf) {  // [B, H]
  __shared__ __attribute__((aligned(16))) __bf16 Ab[BB][APAD];

  const int tid  = threadIdx.x;
  const int wave = tid >> 5;
  const int lane = tid & 31;
  const int j0   = (blockIdx.x * 8 + wave) * 16;

  // async stage hprev: 4096 x 16B chunks, 16 per thread (coalesced)
#pragma unroll
  for (int i = 0; i < 16; ++i) {
    int c = i * 256 + tid;
    int row = c >> 7, col = (c & 127) * 8;
    async_copy_b128(&Ab[row][col], hprev_bf + (size_t)row * HH + col);
  }
  wait_async0();
  __syncthreads();

  v8f acc[3][2] = {};
  const int rowA = lane & 15, halfA = (lane < 16) ? 0 : 8;
  const int rowB = lane & 15, kB    = (lane < 16) ? 0 : 8;

  const __bf16* bb[3];
#pragma unroll
  for (int g = 0; g < 3; ++g)
    bb[g] = WhT + (size_t)(g * HH + j0 + rowB) * HH + kB;

  v16bf bcur[3];
#pragma unroll
  for (int g = 0; g < 3; ++g) bcur[g] = make_frag(bb[g], bb[g] + 16);

  for (int kc = 0; kc < HH; kc += 32) {
    v16bf bnxt[3];
    if (kc + 32 < HH) {
#pragma unroll
      for (int g = 0; g < 3; ++g)
        bnxt[g] = make_frag(bb[g] + kc + 32, bb[g] + kc + 48);
    } else {
#pragma unroll
      for (int g = 0; g < 3; ++g) bnxt[g] = bcur[g];
    }

    v16bf afrag[2];
#pragma unroll
    for (int mt = 0; mt < 2; ++mt) {
      const __bf16* ap = &Ab[mt * 16 + rowA][kc];
      afrag[mt] = make_frag(ap + halfA, ap + 16 + halfA);
    }
#pragma unroll
    for (int g = 0; g < 3; ++g)
#pragma unroll
      for (int mt = 0; mt < 2; ++mt)
        acc[g][mt] = __builtin_amdgcn_wmma_f32_16x16x32_bf16(
            false, afrag[mt], false, bcur[g], (short)0, acc[g][mt],
            false, false);
#pragma unroll
    for (int g = 0; g < 3; ++g) bcur[g] = bnxt[g];
  }

  const int j = j0 + (lane & 15);
#pragma unroll
  for (int mt = 0; mt < 2; ++mt) {
    const int bbase = mt * 16 + ((lane < 16) ? 0 : 8);
#pragma unroll
    for (int r8 = 0; r8 < 8; ++r8) {
      const int b = bbase + r8;
      const float* gxrow = gx + (size_t)b * GG;
      const float rr = sigmoidf_(gxrow[j]      + acc[0][mt][r8]);
      const float zz = sigmoidf_(gxrow[HH + j] + acc[1][mt][r8]);
      const float nn = tanhf(gxrow[2 * HH + j] + rr * acc[2][mt][r8]);
      const float hp = hprev_f[(size_t)b * HH + j];
      const float h  = (1.0f - zz) * nn + zz * hp;
      hnew_f[(size_t)b * HH + j]  = h;
      hnew_bf[(size_t)b * HH + j] = f32_to_bf16_rne(h);
    }
  }
}

// ---------------------------------------------------------------------------
extern "C" void kernel_launch(void* const* d_in, const int* in_sizes, int n_in,
                              void* d_out, int out_size, void* d_ws,
                              size_t ws_size, hipStream_t stream) {
  const float* x   = (const float*)d_in[0];
  const float* Wx0 = (const float*)d_in[1];
  const float* Wh0 = (const float*)d_in[2];
  const float* b0  = (const float*)d_in[3];
  const float* Wx1 = (const float*)d_in[4];
  const float* Wh1 = (const float*)d_in[5];
  const float* b1  = (const float*)d_in[6];
  float* out = (float*)d_out;
  char*  ws  = (char*)d_ws;   // needs ~275 MB

  float*          gx    = (float*)(ws + OFF_GX);
  unsigned short* hall  = (unsigned short*)(ws + OFF_HALL);
  unsigned short* xbf   = (unsigned short*)(ws + OFF_XBF);
  unsigned short* wxt0  = (unsigned short*)(ws + OFF_WXT0);
  unsigned short* wht0  = (unsigned short*)(ws + OFF_WHT0);
  unsigned short* wxt1  = (unsigned short*)(ws + OFF_WXT1);
  unsigned short* wht1  = (unsigned short*)(ws + OFF_WHT1);
  unsigned short* hzero = (unsigned short*)(ws + OFF_HZERO);
  float*          h0f   = (float*)(ws + OFF_H0F);
  unsigned short* h1bf  = (unsigned short*)(ws + OFF_H1BF);
  float*          h1f   = (float*)(ws + OFF_H1F);

  hipMemsetAsync(ws + OFF_STATE, 0, STATE_BYTES, stream);

  transpose_cvt_bf16<<<dim3(GG / 32, II / 32), 256, 0, stream>>>(Wx0, wxt0, II, GG);
  transpose_cvt_bf16<<<dim3(GG / 32, HH / 32), 256, 0, stream>>>(Wh0, wht0, HH, GG);
  transpose_cvt_bf16<<<dim3(GG / 32, HH / 32), 256, 0, stream>>>(Wx1, wxt1, HH, GG);
  transpose_cvt_bf16<<<dim3(GG / 32, HH / 32), 256, 0, stream>>>(Wh1, wht1, HH, GG);
  cvt_x_timemajor<<<(TT * BB * II) / 256, 256, 0, stream>>>(x, xbf);

  const int Mall = TT * BB;  // 16384
  gemm_bf16_bias<<<dim3(GG / 128, Mall / 128), 256, 0, stream>>>(
      (const __bf16*)xbf, (const __bf16*)wxt0, b0, gx, Mall, GG, II);

  for (int t = 0; t < TT; ++t) {
    const __bf16* hp_bf = (t == 0) ? (const __bf16*)hzero
                                   : (const __bf16*)(hall + (size_t)(t - 1) * BB * HH);
    gru_step<<<dim3(HH / 128), 256, 0, stream>>>(
        gx + (size_t)t * BB * GG, hp_bf, h0f + (size_t)(t & 1) * BB * HH,
        (const __bf16*)wht0, h0f + (size_t)((t + 1) & 1) * BB * HH,
        hall + (size_t)t * BB * HH);
  }

  gemm_bf16_bias<<<dim3(GG / 128, Mall / 128), 256, 0, stream>>>(
      (const __bf16*)hall, (const __bf16*)wxt1, b1, gx, Mall, GG, HH);

  for (int t = 0; t < TT; ++t) {
    gru_step<<<dim3(HH / 128), 256, 0, stream>>>(
        gx + (size_t)t * BB * GG,
        (const __bf16*)(h1bf + (size_t)(t & 1) * BB * HH),
        h1f + (size_t)(t & 1) * BB * HH, (const __bf16*)wht1,
        h1f + (size_t)((t + 1) & 1) * BB * HH,
        h1bf + (size_t)((t + 1) & 1) * BB * HH);
  }

  hipMemcpyAsync(out, h1f, (size_t)BB * HH * 4, hipMemcpyDeviceToDevice, stream);
}
